// Mamba2Mixer_67224828117486
// MI455X (gfx1250) — compile-verified, matrix operand order checked by reading, not run on hardware
//
#include <hip/hip_runtime.h>
#include <hip/hip_bf16.h>
#include <math.h>

#define HIDDEN   2048
#define INTER    4096
#define NHEADS   64
#define HEADDIM  64
#define DSTATE   128
#define CONV_DIM 4352
#define PROJ_DIM 8512
#define CHUNK    256
#define NCHUNK   8
#define BATCH    2
#define SEQ      2048
#define ROWS     (BATCH*SEQ)   /* 4096 */

typedef __attribute__((ext_vector_type(16))) __bf16   v16bf;
typedef __attribute__((ext_vector_type(8)))  float    v8f;
typedef __attribute__((ext_vector_type(4)))  float    v4f;
typedef __attribute__((ext_vector_type(2)))  float    v2f;
typedef __attribute__((ext_vector_type(4)))  unsigned u32x4;

union FragBF { v16bf v; u32x4 q[2]; };
union Pack8BF { __bf16 h[8]; u32x4 q; };

static __device__ inline v8f vzero8() {
  v8f z = {0.f,0.f,0.f,0.f,0.f,0.f,0.f,0.f};
  return z;
}

static __device__ inline __bf16 to_bf16(float f) {
  unsigned u = __builtin_bit_cast(unsigned, f);
  u += 0x7FFFu + ((u >> 16) & 1u);          // round-to-nearest-even
  unsigned short hs = (unsigned short)(u >> 16);
  return __builtin_bit_cast(__bf16, hs);
}

static __device__ inline v8f wmma_bf16(v16bf a, v16bf b, v8f c) {
  // D = A(16x32 bf16) * B(32x16 bf16) + C(f32)
  return __builtin_amdgcn_wmma_f32_16x16x32_bf16(false, a, false, b, (short)0, c, false, false);
}

static __device__ inline v8f wmma_f32(v2f a, v2f b, v8f c) {
#if __has_builtin(__builtin_amdgcn_wmma_f32_16x16x4_f32)
  // D = A(16x4 f32) * B(4x16 f32) + C(f32)
  return __builtin_amdgcn_wmma_f32_16x16x4_f32(false, a, false, b, (short)0, c, false, false);
#else
  c[0] += a[0]*b[0] + a[1]*b[1];
  return c;
#endif
}

// CDNA5 async global->LDS copy (ASYNCcnt-tracked), if the toolchain exposes it.
// Parameter type (from clang diagnostic): int __vector(4) __device__* /
// presumably int __vector(4) __shared__* for the LDS side.
#if __has_builtin(__builtin_amdgcn_global_load_async_to_lds_b128)
#define USE_ASYNC_LDS 1
typedef int i32x4v __attribute__((vector_size(16)));
typedef __attribute__((address_space(1))) i32x4v* as1_i32x4p;
typedef __attribute__((address_space(3))) i32x4v* as3_i32x4p;
static __device__ inline void async_copy16(const __bf16* g, __bf16* l) {
  __builtin_amdgcn_global_load_async_to_lds_b128(
      (as1_i32x4p)g, (as3_i32x4p)l, 0, 0);
}
static __device__ inline void wait_async0() {
  asm volatile("s_wait_asynccnt 0x0" ::: "memory");
}
#else
#define USE_ASYNC_LDS 0
#endif

// ---------------------------------------------------------------------------
// fp32 -> bf16 conversion, 8 elems / thread (16B stores)
// ---------------------------------------------------------------------------
__global__ __launch_bounds__(256) void f32_to_bf16_kernel(
    const float* __restrict__ in, __bf16* __restrict__ out, size_t n) {
  size_t i = ((size_t)blockIdx.x * 256 + threadIdx.x) * 8;
  if (i + 8 <= n) {
    v4f x0 = *(const v4f*)(in + i);
    v4f x1 = *(const v4f*)(in + i + 4);
    Pack8BF p;
    p.h[0]=to_bf16(x0.x); p.h[1]=to_bf16(x0.y); p.h[2]=to_bf16(x0.z); p.h[3]=to_bf16(x0.w);
    p.h[4]=to_bf16(x1.x); p.h[5]=to_bf16(x1.y); p.h[6]=to_bf16(x1.z); p.h[7]=to_bf16(x1.w);
    *(u32x4*)(out + i) = p.q;
  } else {
    for (size_t j = i; j < n; ++j) out[j] = to_bf16(in[j]);
  }
}

// ---------------------------------------------------------------------------
// C(MxN, f32) = A(MxK, bf16) * B(NxK, bf16)^T -- WMMA bf16
// block = 256 threads = 8 waves (4x2); block tile 256x128x32; wave tile 64x64
// double-buffered LDS, async global->LDS when available
// M must be a multiple of 256, K a multiple of 32; N guarded at stores
// ---------------------------------------------------------------------------
__global__ __launch_bounds__(256) void gemm_bf16_nt(
    const __bf16* __restrict__ A, const __bf16* __restrict__ B,
    float* __restrict__ C, int M, int N, int K) {
  const int LDT = 40;                          // 32 + 8 bf16 pad (80B row stride)
  __shared__ __bf16 As[2*256*40];              // 40 KB
  __shared__ __bf16 Bs[2*128*40];              // 20 KB

  int tileM = blockIdx.y * 256;
  int tileN = blockIdx.x * 128;
  int tid  = threadIdx.x;
  int lane = tid & 31;
  int w    = tid >> 5;
  int wm   = w >> 1;                           // 0..3 (64-row slab)
  int wn   = w & 1;                            // 0..1 (64-col slab)
  int m16  = lane & 15;
  int hi   = lane >> 4;

  v8f acc[4][4];
  for (int i = 0; i < 4; ++i)
    for (int j = 0; j < 4; ++j) acc[i][j] = vzero8();

  int arow = tid;                              // 0..255: full 32-K row of A (64B)
  int brow = tid >> 1;                         // 0..127
  int bhal = tid & 1;                          // 32B half of the 32-K row of B

  const __bf16* Ap = A + (size_t)(tileM + arow) * K;
  int gb = tileN + brow; if (gb > N - 1) gb = N - 1;   // clamp: no exec branches
  const __bf16* Bp = B + (size_t)gb * K;

  int nk = K / 32;

  // ---- preload tile 0 into buffer 0 ----
#if USE_ASYNC_LDS
  for (int j = 0; j < 4; ++j)
    async_copy16(Ap + j*8, As + arow*LDT + j*8);
  for (int j = 0; j < 2; ++j)
    async_copy16(Bp + bhal*16 + j*8, Bs + brow*LDT + bhal*16 + j*8);
  wait_async0();
#else
  {
    u32x4 ra[4], rb[2];
    for (int j = 0; j < 4; ++j) ra[j] = *(const u32x4*)(Ap + j*8);
    for (int j = 0; j < 2; ++j) rb[j] = *(const u32x4*)(Bp + bhal*16 + j*8);
    u32x4* da = (u32x4*)(As + arow*LDT);
    for (int j = 0; j < 4; ++j) da[j] = ra[j];
    u32x4* db = (u32x4*)(Bs + brow*LDT + bhal*16);
    for (int j = 0; j < 2; ++j) db[j] = rb[j];
  }
#endif
  __syncthreads();

  for (int kt = 0; kt < nk; ++kt) {
    int cur = kt & 1, nxt = cur ^ 1;
    bool have_next = (kt + 1 < nk);
    int k1 = (kt + 1) * 32;

#if USE_ASYNC_LDS
    if (have_next) {
      for (int j = 0; j < 4; ++j)
        async_copy16(Ap + k1 + j*8, As + nxt*256*LDT + arow*LDT + j*8);
      for (int j = 0; j < 2; ++j)
        async_copy16(Bp + k1 + bhal*16 + j*8,
                     Bs + nxt*128*LDT + brow*LDT + bhal*16 + j*8);
    }
#else
    u32x4 ra[4], rb[2];
    if (have_next) {
      for (int j = 0; j < 4; ++j) ra[j] = *(const u32x4*)(Ap + k1 + j*8);
      for (int j = 0; j < 2; ++j) rb[j] = *(const u32x4*)(Bp + k1 + bhal*16 + j*8);
    }
#endif

    // ---- compute from buffer `cur` ----
    const __bf16* Ab = As + cur*256*LDT;
    const __bf16* Bb = Bs + cur*128*LDT;
    FragBF fa[4], fb[4];
    // ISA 16-bit A 16x32 frag: elems 0..7 <- K=hi*8.., elems 8..15 <- K=16+hi*8..
    for (int mi = 0; mi < 4; ++mi) {
      int r = wm*64 + mi*16 + m16;
      fa[mi].q[0] = *(const u32x4*)(Ab + r*LDT + hi*8);
      fa[mi].q[1] = *(const u32x4*)(Ab + r*LDT + 16 + hi*8);
    }
    for (int ni = 0; ni < 4; ++ni) {
      int r = wn*64 + ni*16 + m16;
      fb[ni].q[0] = *(const u32x4*)(Bb + r*LDT + hi*8);
      fb[ni].q[1] = *(const u32x4*)(Bb + r*LDT + 16 + hi*8);
    }
    for (int mi = 0; mi < 4; ++mi)
      for (int ni = 0; ni < 4; ++ni)
        acc[mi][ni] = wmma_bf16(fa[mi].v, fb[ni].v, acc[mi][ni]);

#if USE_ASYNC_LDS
    wait_async0();
#else
    if (have_next) {
      u32x4* da = (u32x4*)(As + nxt*256*LDT + arow*LDT);
      for (int j = 0; j < 4; ++j) da[j] = ra[j];
      u32x4* db = (u32x4*)(Bs + nxt*128*LDT + brow*LDT + bhal*16);
      for (int j = 0; j < 2; ++j) db[j] = rb[j];
    }
#endif
    __syncthreads();
  }

  // C/D layout: VGPR r -> row = r + hi*8, col = m16
  for (int mi = 0; mi < 4; ++mi)
    for (int ni = 0; ni < 4; ++ni)
      for (int r = 0; r < 8; ++r) {
        int row = tileM + wm*64 + mi*16 + hi*8 + r;
        int col = tileN + wn*64 + ni*16 + m16;
        if (col < N)
          C[(size_t)row * N + col] = acc[mi][ni][r];
      }
}

// ---------------------------------------------------------------------------
// depthwise causal conv (K=4) + SiLU over the hbc slice of proj
// ---------------------------------------------------------------------------
__global__ __launch_bounds__(256) void conv_silu_kernel(
    const float* __restrict__ proj, const float* __restrict__ cw,
    const float* __restrict__ cb, float* __restrict__ xbc) {
  size_t idx = (size_t)blockIdx.x * 256 + threadIdx.x;
  if (idx >= (size_t)ROWS * CONV_DIM) return;
  int ccol = (int)(idx % CONV_DIM);
  size_t row = idx / CONV_DIM;               // b*SEQ + s
  int s = (int)(row & (SEQ - 1));
  float v = cb[ccol];
  for (int k = 0; k < 4; ++k) {
    int sp = s - 3 + k;
    if (sp >= 0)
      v += proj[(row - 3 + k) * PROJ_DIM + INTER + ccol] * cw[ccol*4 + k];
  }
  xbc[idx] = v / (1.f + __expf(-v));         // silu
}

// ---------------------------------------------------------------------------
// dt = softplus(proj_dt + dt_bias); a = dt * (-exp(A_log))
// ---------------------------------------------------------------------------
__global__ __launch_bounds__(256) void dt_kernel(
    const float* __restrict__ proj, const float* __restrict__ dt_bias,
    const float* __restrict__ A_log, float* __restrict__ dtA,
    float* __restrict__ aA) {
  int idx = blockIdx.x * 256 + threadIdx.x;
  if (idx >= ROWS * NHEADS) return;
  int hh = idx & 63;
  int row = idx >> 6;
  float r = proj[(size_t)row * PROJ_DIM + (INTER + CONV_DIM) + hh] + dt_bias[hh];
  float dt = (r > 20.f) ? r : log1pf(__expf(r));
  dtA[idx] = dt;
  aA[idx]  = -dt * __expf(A_log[hh]);
}

// ---------------------------------------------------------------------------
// scan A: per (b,chunk,head): cumsum(a), Y_diag = (C B^T . expmask) (x*dt),
// states = (x*dt*decay)^T B.  128 threads = 4 waves, f32 WMMA 16x16x4.
// ---------------------------------------------------------------------------
__global__ __launch_bounds__(128) void scan_chunk_kernel(
    const float* __restrict__ xbc, const float* __restrict__ dtA,
    const float* __restrict__ aArr, float* __restrict__ y,
    float* __restrict__ states, float* __restrict__ cdec) {
  extern __shared__ float sm[];
  const int LXT = 258;                       // padded stride (even, 8B ok)
  float* cs_s  = sm;                         // 256
  float* dec_s = sm + 256;                   // 256
  float* XT    = sm + 512;                   // 64 * 258  (XT[p][s] = x*dt, transposed)
  float* strip = XT + 64*LXT;                // 4 waves * 16 * 258 (Gm strips)

  int bidx = blockIdx.x;
  int h = bidx & 63;
  int c = (bidx >> 6) & 7;
  int b = bidx >> 9;
  int tid = threadIdx.x;
  int lane = tid & 31;
  int w = tid >> 5;
  int m16 = lane & 15, hi = lane >> 4;
  size_t rowbase = (size_t)b * SEQ + (size_t)c * CHUNK;

  for (int i = tid; i < 256; i += 128)
    cs_s[i] = aArr[(rowbase + i) * NHEADS + h];
  __syncthreads();
  if (tid == 0) { float a = 0.f; for (int l = 0; l < 256; ++l) { a += cs_s[l]; cs_s[l] = a; } }
  __syncthreads();
  float cs_last = cs_s[255];
  for (int i = tid; i < 256; i += 128)
    dec_s[i] = __expf(cs_last - cs_s[i]);
  if (tid == 0) cdec[bidx] = __expf(cs_last);

  for (int i = tid; i < CHUNK * HEADDIM; i += 128) {
    int s = i >> 6, p = i & 63;
    float xv = xbc[(rowbase + s) * CONV_DIM + h*HEADDIM + p];
    XT[p*LXT + s] = xv * dtA[(rowbase + s) * NHEADS + h];
  }
  __syncthreads();

  float* mystrip = strip + w * 16 * LXT;
  for (int mt = w*4; mt < w*4 + 4; ++mt) {
    // G strip (16 x 256) = C-rows (16 x 128) @ B^T, contraction over n
    v8f acc[16];
    for (int i = 0; i < 16; ++i) acc[i] = vzero8();
    const float* Crow = xbc + (rowbase + mt*16 + m16) * CONV_DIM + INTER + DSTATE;
    for (int kk = 0; kk < 32; ++kk) {
      int k0 = kk*4 + hi*2;                  // n index pair for this lane
      v2f af = *(const v2f*)(Crow + k0);     // contiguous
      for (int st = 0; st < 16; ++st) {
        const float* Brow = xbc + (rowbase + st*16 + m16) * CONV_DIM + INTER;
        v2f bf_ = *(const v2f*)(Brow + k0);  // B^T operand: contiguous along n
        acc[st] = wmma_f32(af, bf_, acc[st]);
      }
    }
    // causal exp mask, spill strip to LDS (per-wave region)
    for (int st = 0; st < 16; ++st)
      for (int r = 0; r < 8; ++r) {
        int l = mt*16 + hi*8 + r;
        int s = st*16 + m16;
        float v = acc[st][r];
        v = (s <= l) ? v * __expf(cs_s[l] - cs_s[s]) : 0.f;
        mystrip[(hi*8 + r) * LXT + s] = v;
      }
    // Y_diag tile (16 x 64) = strip (16 x 256) @ X (256 x 64), contraction over s
    v8f yacc[4];
    for (int i = 0; i < 4; ++i) yacc[i] = vzero8();
    for (int kk = 0; kk < 64; ++kk) {
      int k0 = kk*4 + hi*2;                  // s index pair
      v2f af = *(const v2f*)(mystrip + m16*LXT + k0);
      for (int nt = 0; nt < 4; ++nt) {
        v2f bf_ = *(const v2f*)(XT + (nt*16 + m16) * LXT + k0); // X^T contiguous along s
        yacc[nt] = wmma_f32(af, bf_, yacc[nt]);
      }
    }
    for (int nt = 0; nt < 4; ++nt)
      for (int r = 0; r < 8; ++r) {
        int l = mt*16 + hi*8 + r;
        y[(rowbase + l) * INTER + h*HEADDIM + nt*16 + m16] = yacc[nt][r];
      }
  }
  __syncthreads();

  // states (64 x 128) = (X * decay)^T (64 x 256) @ B (256 x 128)
  for (int pt = 0; pt < 4; ++pt)
    for (int nt2 = 2*w; nt2 < 2*w + 2; ++nt2) {
      v8f sacc = vzero8();
      for (int kk = 0; kk < 64; ++kk) {
        int k0 = kk*4 + hi*2;                // s index pair
        v2f t = *(const v2f*)(XT + (pt*16 + m16) * LXT + k0);
        v2f af = { t.x * dec_s[k0], t.y * dec_s[k0 + 1] };
        int n = nt2*16 + m16;
        v2f bf_ = { xbc[(rowbase + k0    ) * CONV_DIM + INTER + n],
                    xbc[(rowbase + k0 + 1) * CONV_DIM + INTER + n] };
        sacc = wmma_f32(af, bf_, sacc);
      }
      for (int r = 0; r < 8; ++r) {
        int p = pt*16 + hi*8 + r;
        int n = nt2*16 + m16;
        states[(size_t)bidx * (HEADDIM*DSTATE) + p*DSTATE + n] = sacc[r];
      }
    }
}

// ---------------------------------------------------------------------------
// scan B: sequential carry over chunks; prev[c] = carry before chunk c
// ---------------------------------------------------------------------------
__global__ __launch_bounds__(256) void scan_carry_kernel(
    const float* __restrict__ states, const float* __restrict__ cdec,
    float* __restrict__ prev) {
  int bh = blockIdx.x;                       // b*64 + h
  int b = bh >> 6, h = bh & 63;
  int tid = threadIdx.x;
  float carry[32];
  for (int i = 0; i < 32; ++i) carry[i] = 0.f;
  for (int c = 0; c < NCHUNK; ++c) {
    size_t base = ((size_t)(b*NCHUNK + c) * NHEADS + h) * (HEADDIM*DSTATE);
    float d = cdec[(b*NCHUNK + c) * NHEADS + h];
    for (int i = 0; i < 32; ++i) {
      int idx = tid + i*256;
      prev[base + idx] = carry[i];
      carry[i] = carry[i] * d + states[base + idx];
    }
  }
}

// ---------------------------------------------------------------------------
// scan C: y += exp(cs) * (C @ prev^T) + D*x   (f32 WMMA, contraction over n)
// ---------------------------------------------------------------------------
__global__ __launch_bounds__(128) void scan_yoff_kernel(
    const float* __restrict__ xbc, const float* __restrict__ aArr,
    const float* __restrict__ prev, const float* __restrict__ Dvec,
    float* __restrict__ y) {
  __shared__ float cs_s[256];
  __shared__ float exps[256];
  int bidx = blockIdx.x;
  int h = bidx & 63;
  int c = (bidx >> 6) & 7;
  int b = bidx >> 9;
  int tid = threadIdx.x;
  int lane = tid & 31;
  int w = tid >> 5;
  int m16 = lane & 15, hi = lane >> 4;
  size_t rowbase = (size_t)b * SEQ + (size_t)c * CHUNK;

  for (int i = tid; i < 256; i += 128)
    cs_s[i] = aArr[(rowbase + i) * NHEADS + h];
  __syncthreads();
  if (tid == 0) { float a = 0.f; for (int l = 0; l < 256; ++l) { a += cs_s[l]; cs_s[l] = a; } }
  __syncthreads();
  for (int i = tid; i < 256; i += 128) exps[i] = __expf(cs_s[i]);
  __syncthreads();

  float Dh = Dvec[h];
  size_t pbase = (size_t)bidx * (HEADDIM*DSTATE);

  for (int mt = w*4; mt < w*4 + 4; ++mt) {
    v8f acc[4];
    for (int i = 0; i < 4; ++i) acc[i] = vzero8();
    const float* Crow = xbc + (rowbase + mt*16 + m16) * CONV_DIM + INTER + DSTATE;
    for (int kk = 0; kk < 32; ++kk) {
      int k0 = kk*4 + hi*2;                  // n index pair
      v2f af = *(const v2f*)(Crow + k0);
      for (int pt = 0; pt < 4; ++pt) {
        int p = pt*16 + m16;
        v2f bf_ = *(const v2f*)(prev + pbase + (size_t)p*DSTATE + k0); // prev^T contiguous
        acc[pt] = wmma_f32(af, bf_, acc[pt]);
      }
    }
    for (int pt = 0; pt < 4; ++pt)
      for (int r = 0; r < 8; ++r) {
        int l = mt*16 + hi*8 + r;
        int p = pt*16 + m16;
        size_t row = rowbase + l;
        size_t oidx = row * INTER + h*HEADDIM + p;
        float xv = xbc[row * CONV_DIM + h*HEADDIM + p];
        y[oidx] += acc[pt][r] * exps[l] + Dh * xv;
      }
  }
}

// ---------------------------------------------------------------------------
// gated RMSNorm: h = y * silu(gate); out_bf16 = nw * h * rsqrt(mean(h^2)+eps)
// ---------------------------------------------------------------------------
__global__ __launch_bounds__(256) void norm_kernel(
    const float* __restrict__ y, const float* __restrict__ proj,
    const float* __restrict__ nw, __bf16* __restrict__ out) {
  int row = blockIdx.x;
  int tid = threadIdx.x;
  __shared__ float red[256];
  float hv[16];
  float ss = 0.f;
  for (int i = 0; i < 16; ++i) {
    int col = tid + i*256;
    float g = proj[(size_t)row * PROJ_DIM + col];     // gate slice
    float sg = g / (1.f + __expf(-g));
    float v = y[(size_t)row * INTER + col] * sg;
    hv[i] = v; ss += v * v;
  }
  red[tid] = ss;
  __syncthreads();
  for (int o = 128; o > 0; o >>= 1) {
    if (tid < o) red[tid] += red[tid + o];
    __syncthreads();
  }
  float inv = rsqrtf(red[0] / (float)INTER + 1e-6f);
  for (int i = 0; i < 16; ++i) {
    int col = tid + i*256;
    out[(size_t)row * INTER + col] = to_bf16(nw[col] * hv[i] * inv);
  }
}

// ---------------------------------------------------------------------------
extern "C" void kernel_launch(void* const* d_in, const int* in_sizes, int n_in,
                              void* d_out, int out_size, void* d_ws, size_t ws_size,
                              hipStream_t stream) {
  (void)in_sizes; (void)n_in; (void)out_size;
  const float* hidden  = (const float*)d_in[0];
  const float* w_in    = (const float*)d_in[1];
  const float* conv_w  = (const float*)d_in[2];
  const float* conv_b  = (const float*)d_in[3];
  const float* dt_bias = (const float*)d_in[4];
  const float* A_log   = (const float*)d_in[5];
  const float* Dvec    = (const float*)d_in[6];
  const float* norm_w  = (const float*)d_in[7];
  const float* w_out   = (const float*)d_in[8];
  float* out = (float*)d_out;

  char* ws = (char*)d_ws;
  size_t off = 0;
  auto alloc = [&](size_t bytes) -> char* {
    char* p = ws + off;
    off += (bytes + 255) & ~(size_t)255;
    return p;
  };
  __bf16* hsb    = (__bf16*)alloc((size_t)ROWS * HIDDEN * 2);
  __bf16* w1b    = (__bf16*)alloc((size_t)PROJ_DIM * HIDDEN * 2);
  __bf16* w2b    = (__bf16*)alloc((size_t)HIDDEN * INTER * 2);
  float*  proj   = (float*)alloc((size_t)ROWS * PROJ_DIM * 4);
  float*  xbc    = (float*)alloc((size_t)ROWS * CONV_DIM * 4);
  float*  dtA    = (float*)alloc((size_t)ROWS * NHEADS * 4);
  float*  aA     = (float*)alloc((size_t)ROWS * NHEADS * 4);
  float*  ybuf   = (float*)alloc((size_t)ROWS * INTER * 4);
  float*  states = (float*)alloc((size_t)BATCH*NCHUNK*NHEADS * HEADDIM*DSTATE * 4);
  float*  prev   = (float*)alloc((size_t)BATCH*NCHUNK*NHEADS * HEADDIM*DSTATE * 4);
  float*  cdec   = (float*)alloc((size_t)BATCH*NCHUNK*NHEADS * 4);
  __bf16* hnb    = (__bf16*)alloc((size_t)ROWS * INTER * 2);
  if (off > ws_size) return;                 // workspace too small: bail safely

  size_t nh = (size_t)ROWS * HIDDEN;
  f32_to_bf16_kernel<<<(unsigned)((nh/8 + 255) / 256), 256, 0, stream>>>(hidden, hsb, nh);
  size_t nw1 = (size_t)PROJ_DIM * HIDDEN;
  f32_to_bf16_kernel<<<(unsigned)((nw1/8 + 255) / 256), 256, 0, stream>>>(w_in, w1b, nw1);
  size_t nw2 = (size_t)HIDDEN * INTER;
  f32_to_bf16_kernel<<<(unsigned)((nw2/8 + 255) / 256), 256, 0, stream>>>(w_out, w2b, nw2);

  dim3 g1((PROJ_DIM + 127) / 128, ROWS / 256);
  gemm_bf16_nt<<<g1, 256, 0, stream>>>(hsb, w1b, proj, ROWS, PROJ_DIM, HIDDEN);

  size_t nconv = (size_t)ROWS * CONV_DIM;
  conv_silu_kernel<<<(unsigned)((nconv + 255) / 256), 256, 0, stream>>>(proj, conv_w, conv_b, xbc);

  dt_kernel<<<(ROWS * NHEADS) / 256, 256, 0, stream>>>(proj, dt_bias, A_log, dtA, aA);

  const unsigned scan_smem = (256 + 256 + 64*258 + 4*16*258) * sizeof(float); // ~131 KB
  scan_chunk_kernel<<<BATCH*NCHUNK*NHEADS, 128, scan_smem, stream>>>(
      xbc, dtA, aA, ybuf, states, cdec);
  scan_carry_kernel<<<BATCH*NHEADS, 256, 0, stream>>>(states, cdec, prev);
  scan_yoff_kernel<<<BATCH*NCHUNK*NHEADS, 128, 0, stream>>>(xbc, aA, prev, Dvec, ybuf);

  norm_kernel<<<ROWS, 256, 0, stream>>>(ybuf, proj, norm_w, hnb);

  dim3 g2(HIDDEN / 128, ROWS / 256);
  gemm_bf16_nt<<<g2, 256, 0, stream>>>(hnb, w2b, out, ROWS, HIDDEN, INTER);
}
